// Critic_74423193305351
// MI455X (gfx1250) — compile-verified, matrix operand order checked by reading, not run on hardware
//
#include <hip/hip_runtime.h>

// ---------------------------------------------------------------------------
// Problem constants (match reference)
// ---------------------------------------------------------------------------
#define NN      32768
#define EE      524288
#define BG      256
#define HID     64
#define NLAYER  3
#define STATE_DIM 16
#define ACT_DIM   2
#define CAT_DIM  (HID + STATE_DIM + ACT_DIM)   // 82
#define MLP1    256
#define MLP2    256
#define EPS_AGG 1e-6f
#define EPS_BN  1e-5f

typedef __bf16 bf16_t;
typedef bf16_t v16bf  __attribute__((ext_vector_type(16)));
typedef bf16_t bf16x4 __attribute__((ext_vector_type(4)));
typedef float  v8f    __attribute__((ext_vector_type(8)));

// ---------------------------------------------------------------------------
// Fragment builder. `s` is an LDS tile indexed [m][k] with k-stride 64
// (A tile: [row][k]; W tile stored transposed: [col][k]).  Builds the two
// K=32 bf16 fragments per ISA 16-bit A-matrix 16x32 layout:
//   lanes 0-15:  M=lane,    K = {0..7, 16..23}
//   lanes 16-31: M=lane-16, K = {8..15, 24..31}
// ---------------------------------------------------------------------------
__device__ __forceinline__ void build_frag_pair(const bf16_t (*s)[HID],
                                                int base16, int lane,
                                                v16bf& f0, v16bf& f1)
{
    const int r  = base16 + (lane & 15);
    const int kb = (lane < 16) ? 0 : 8;
    #pragma unroll
    for (int j = 0; j < 16; ++j) {
        const int kk = kb + ((j < 8) ? j : (j + 8));
        f0[j] = s[r][kk];
        f1[j] = s[r][32 + kk];
    }
}

// Stage a 32x64 fp32 tile into LDS as bf16: 2 unconditional float4 loads
// per thread (trip count is compile-time; avoids EXEC-masked staging).
__device__ __forceinline__ void stage_A32(bf16_t (*sA)[HID],
                                          const float* __restrict__ Xtile, int tid)
{
    const float4* Xv = (const float4*)Xtile;   // 512 float4
    #pragma unroll
    for (int it = 0; it < 2; ++it) {
        const int f = tid + it * 256;
        const float4 v = Xv[f];
        const int r = f >> 4, c4 = (f & 15) << 2;
        bf16x4 p = { (bf16_t)v.x, (bf16_t)v.y, (bf16_t)v.z, (bf16_t)v.w };
        *(bf16x4*)&sA[r][c4] = p;
    }
}

// Stage a 64x64 fp32 weight matrix into LDS transposed ([col][k]) as bf16.
__device__ __forceinline__ void stage_W64(bf16_t (*sW)[HID],
                                          const float* __restrict__ W, int tid)
{
    const float4* Wv = (const float4*)W;       // 1024 float4
    #pragma unroll
    for (int it = 0; it < 4; ++it) {
        const int f = tid + it * 256;
        const float4 v = Wv[f];
        const int k = f >> 4, c4 = (f & 15) << 2;
        sW[c4 + 0][k] = (bf16_t)v.x;
        sW[c4 + 1][k] = (bf16_t)v.y;
        sW[c4 + 2][k] = (bf16_t)v.z;
        sW[c4 + 3][k] = (bf16_t)v.w;
    }
}

// ---------------------------------------------------------------------------
// Specialized K=64 -> Nout=64 GEMM (W resident in LDS, grid-stride row tiles)
// ---------------------------------------------------------------------------
__global__ __launch_bounds__(256)
void gemm64_wmma(const float* __restrict__ X, const float* __restrict__ W,
                 const float* __restrict__ bias, float* __restrict__ Y, int M)
{
    __shared__ bf16_t sW[HID][HID];
    __shared__ bf16_t sA[32][HID];

    const int tid  = threadIdx.x;
    const int lane = tid & 31;
    const int wid  = tid >> 5;
    const int rt   = wid >> 2;
    const int ct   = wid & 3;

    stage_W64(sW, W, tid);
    __syncthreads();

    v16bf bf0, bf1;
    build_frag_pair(sW, ct * 16, lane, bf0, bf1);

    const int col  = ct * 16 + (lane & 15);
    const int rOff = (lane < 16) ? 0 : 8;
    const float bv = bias[col];
    const int step = gridDim.x * 32;

    for (int rowBase = blockIdx.x * 32; rowBase < M; rowBase += step) {
        __syncthreads();
        stage_A32(sA, X + (long long)rowBase * HID, tid);
        if (rowBase + step < M) {
            const char* nx = (const char*)(X + (long long)(rowBase + step) * HID);
            __builtin_prefetch(nx + tid * 16, 0, 1);
            __builtin_prefetch(nx + 4096 + tid * 16, 0, 1);
        }
        __syncthreads();

        v16bf af0, af1;
        build_frag_pair(sA, rt * 16, lane, af0, af1);
        v8f acc = {};
        acc = __builtin_amdgcn_wmma_f32_16x16x32_bf16(false, af0, false, bf0,
                                                      (short)0, acc, false, false);
        acc = __builtin_amdgcn_wmma_f32_16x16x32_bf16(false, af1, false, bf1,
                                                      (short)0, acc, false, false);

        float* yp = Y + (long long)(rowBase + rt * 16 + rOff) * HID + col;
        #pragma unroll
        for (int r8 = 0; r8 < 8; ++r8) yp[r8 * HID] = acc[r8] + bv;
    }
}

// ---------------------------------------------------------------------------
// Fused node GEMMs: Ah/Bh/Dh/Eh = h @ {A,B,D,E}_W + bias  (shared A operand)
// ---------------------------------------------------------------------------
__global__ __launch_bounds__(256)
void node_gemm4_wmma(const float* __restrict__ X,
                     const float* __restrict__ Wa, const float* __restrict__ ba,
                     const float* __restrict__ Wb, const float* __restrict__ bb,
                     const float* __restrict__ Wd, const float* __restrict__ bd,
                     const float* __restrict__ We, const float* __restrict__ be,
                     float* __restrict__ Ya, float* __restrict__ Yb,
                     float* __restrict__ Yd, float* __restrict__ Ye, int M)
{
    __shared__ bf16_t sW[4][HID][HID];   // 32 KB
    __shared__ bf16_t sA[32][HID];       // 4 KB

    const int tid  = threadIdx.x;
    const int lane = tid & 31;
    const int wid  = tid >> 5;
    const int rt   = wid >> 2;
    const int ct   = wid & 3;

    stage_W64(sW[0], Wa, tid);
    stage_W64(sW[1], Wb, tid);
    stage_W64(sW[2], Wd, tid);
    stage_W64(sW[3], We, tid);
    __syncthreads();

    v16bf bf[4][2];
    #pragma unroll
    for (int m = 0; m < 4; ++m)
        build_frag_pair(sW[m], ct * 16, lane, bf[m][0], bf[m][1]);

    const int col  = ct * 16 + (lane & 15);
    const int rOff = (lane < 16) ? 0 : 8;
    const float bv[4] = {ba[col], bb[col], bd[col], be[col]};
    float* Ys[4] = {Ya, Yb, Yd, Ye};
    const int step = gridDim.x * 32;

    for (int rowBase = blockIdx.x * 32; rowBase < M; rowBase += step) {
        __syncthreads();
        stage_A32(sA, X + (long long)rowBase * HID, tid);
        if (rowBase + step < M) {
            const char* nx = (const char*)(X + (long long)(rowBase + step) * HID);
            __builtin_prefetch(nx + tid * 16, 0, 1);
            __builtin_prefetch(nx + 4096 + tid * 16, 0, 1);
        }
        __syncthreads();

        v16bf af0, af1;
        build_frag_pair(sA, rt * 16, lane, af0, af1);
        const long long rbase = (long long)(rowBase + rt * 16 + rOff) * HID + col;
        #pragma unroll
        for (int m = 0; m < 4; ++m) {
            v8f acc = {};
            acc = __builtin_amdgcn_wmma_f32_16x16x32_bf16(false, af0, false, bf[m][0],
                                                          (short)0, acc, false, false);
            acc = __builtin_amdgcn_wmma_f32_16x16x32_bf16(false, af1, false, bf[m][1],
                                                          (short)0, acc, false, false);
            float* yp = Ys[m] + rbase;
            #pragma unroll
            for (int r8 = 0; r8 < 8; ++r8) yp[r8 * HID] = acc[r8] + bv[m];
        }
    }
}

// ---------------------------------------------------------------------------
// Fused edge pipeline: Ce GEMM + gather (Dh[src], Eh[dst]) + sigmoid +
// atomic segment-sum (num/den) + BN sum/sumsq accumulation in the epilogue.
// ---------------------------------------------------------------------------
__global__ __launch_bounds__(256)
void edge_gemm_fused(const float* __restrict__ X,
                     const float* __restrict__ W, const float* __restrict__ bias,
                     const int* __restrict__ src, const int* __restrict__ dst,
                     const float* __restrict__ Dh, const float* __restrict__ Eh,
                     const float* __restrict__ Bh,
                     float* __restrict__ ehat,
                     float* __restrict__ num, float* __restrict__ den,
                     float* __restrict__ sums, float* __restrict__ sumsq, int M)
{
    __shared__ bf16_t sW[HID][HID];
    __shared__ bf16_t sA[32][HID];

    const int tid  = threadIdx.x;
    const int lane = tid & 31;
    const int wid  = tid >> 5;
    const int rt   = wid >> 2;
    const int ct   = wid & 3;

    stage_W64(sW, W, tid);
    __syncthreads();

    v16bf bf0, bf1;
    build_frag_pair(sW, ct * 16, lane, bf0, bf1);

    const int col  = ct * 16 + (lane & 15);
    const int rOff = (lane < 16) ? 0 : 8;
    const float bv = bias[col];
    const int step = gridDim.x * 32;

    for (int rowBase = blockIdx.x * 32; rowBase < M; rowBase += step) {
        __syncthreads();
        stage_A32(sA, X + (long long)rowBase * HID, tid);
        if (rowBase + step < M) {
            const char* nx = (const char*)(X + (long long)(rowBase + step) * HID);
            __builtin_prefetch(nx + tid * 16, 0, 1);
            __builtin_prefetch(nx + 4096 + tid * 16, 0, 1);
        }
        __syncthreads();

        v16bf af0, af1;
        build_frag_pair(sA, rt * 16, lane, af0, af1);
        v8f acc = {};
        acc = __builtin_amdgcn_wmma_f32_16x16x32_bf16(false, af0, false, bf0,
                                                      (short)0, acc, false, false);
        acc = __builtin_amdgcn_wmma_f32_16x16x32_bf16(false, af1, false, bf1,
                                                      (short)0, acc, false, false);

        const int rowB = rowBase + rt * 16 + rOff;
        const int* srcp = src + rowB;
        const int* dstp = dst + rowB;
        float* ep = ehat + (long long)rowB * HID + col;

        float s = 0.f, q = 0.f;
        #pragma unroll
        for (int r8 = 0; r8 < 8; ++r8) {
            const int sI = srcp[r8], dI = dstp[r8];
            const float v = acc[r8] + bv + Dh[(long long)sI * HID + col]
                                         + Eh[(long long)dI * HID + col];
            ep[r8 * HID] = v;
            const float sig = 1.0f / (1.0f + __expf(-v));
            atomicAdd(&num[(long long)dI * HID + col],
                      sig * Bh[(long long)sI * HID + col]);
            atomicAdd(&den[(long long)dI * HID + col], sig);
            s += v; q += v * v;
        }
        atomicAdd(&sums[col], s);
        atomicAdd(&sumsq[col], q);
    }
}

// ---------------------------------------------------------------------------
// Generic tiled WMMA GEMM for odd K (embeddings, MLP): Y = act(X@W + bias)
// ---------------------------------------------------------------------------
#define TILE_M 32
#define TILE_N 64
#define TILE_K 32

__global__ __launch_bounds__(256)
void gemm_bias_wmma(const float* __restrict__ X, const float* __restrict__ W,
                    const float* __restrict__ bias, float* __restrict__ Y,
                    int M, int K, int Nout, int relu, int recipX)
{
    __shared__ bf16_t sA[TILE_M][TILE_K];
    __shared__ bf16_t sB[TILE_N][TILE_K];

    const int tid  = threadIdx.x;
    const int lane = tid & 31;
    const int wid  = tid >> 5;
    const int rt   = wid >> 2;
    const int ct   = wid & 3;

    const int rowBase = blockIdx.x * TILE_M;
    const int colBase = blockIdx.y * TILE_N;

    v8f acc = {};
    const int nChunks = (K + TILE_K - 1) / TILE_K;
    for (int kc = 0; kc < nChunks; ++kc) {
        const int k0 = kc * TILE_K;
        // A chunk: 1024 elems -> 4 unconditional elems per thread
        #pragma unroll
        for (int it = 0; it < 4; ++it) {
            const int i = tid + it * 256;
            const int r = i >> 5, k = i & 31;
            const int gr = rowBase + r, gk = k0 + k;
            float v = 0.f;
            if (gr < M && gk < K) {
                v = X[(long long)gr * K + gk];
                if (recipX) v = 1.0f / v;
            }
            sA[r][k] = (bf16_t)v;
        }
        // B chunk transposed: 2048 elems -> 8 unconditional elems per thread
        #pragma unroll
        for (int it = 0; it < 8; ++it) {
            const int i = tid + it * 256;
            const int k = i >> 6, c = i & 63;
            const int gk = k0 + k, gc = colBase + c;
            float v = 0.f;
            if (gk < K) v = W[(long long)gk * Nout + gc];
            sB[c][k] = (bf16_t)v;
        }
        __syncthreads();

        v16bf afrag, bfrag;
        const int rr = lane & 15;
        const int kb = (lane < 16) ? 0 : 8;
        #pragma unroll
        for (int j = 0; j < 16; ++j) {
            const int kk = kb + ((j < 8) ? j : (j + 8));
            afrag[j] = sA[rt * 16 + rr][kk];
            bfrag[j] = sB[ct * 16 + rr][kk];
        }
        acc = __builtin_amdgcn_wmma_f32_16x16x32_bf16(false, afrag, false, bfrag,
                                                      (short)0, acc, false, false);
        __syncthreads();
    }

    const int col  = colBase + ct * 16 + (lane & 15);
    const int rOff = (lane < 16) ? 0 : 8;
    const float bv = bias[col];
    #pragma unroll
    for (int r8 = 0; r8 < 8; ++r8) {
        const int row = rowBase + rt * 16 + r8 + rOff;
        if (row < M) {
            float y = acc[r8] + bv;
            if (relu) y = fmaxf(y, 0.f);
            Y[(long long)row * Nout + col] = y;
        }
    }
}

// ---------------------------------------------------------------------------
// Elementwise / reduction helpers (float4 vectorized)
// ---------------------------------------------------------------------------
__global__ void zero4_kernel(float4* __restrict__ p, long long n4)
{
    long long i = (long long)blockIdx.x * blockDim.x + threadIdx.x;
    const long long stride = (long long)gridDim.x * blockDim.x;
    const float4 z = {0.f, 0.f, 0.f, 0.f};
    for (; i < n4; i += stride) p[i] = z;
}

__global__ void node_update_kernel(const float4* __restrict__ Ah,
                                   const float4* __restrict__ num,
                                   const float4* __restrict__ den,
                                   float4* __restrict__ hnew, long long n4)
{
    const long long i = (long long)blockIdx.x * blockDim.x + threadIdx.x;
    if (i >= n4) return;
    const float4 a = Ah[i], nu = num[i], de = den[i];
    float4 r;
    r.x = a.x + nu.x / (de.x + EPS_AGG);
    r.y = a.y + nu.y / (de.y + EPS_AGG);
    r.z = a.z + nu.z / (de.z + EPS_AGG);
    r.w = a.w + nu.w / (de.w + EPS_AGG);
    hnew[i] = r;
}

#define STAT_ROWS 512
__global__ __launch_bounds__(256)
void stats_kernel(const float* __restrict__ X, long long M,
                  float* __restrict__ sums, float* __restrict__ sumsq)
{
    __shared__ float sS[256], sQ[256];
    const int tid = threadIdx.x;
    const int c   = tid & 63;
    const int rl  = tid >> 6;
    const long long r0 = (long long)blockIdx.x * STAT_ROWS;
    float s = 0.f, q = 0.f;
    for (long long r = r0 + rl; r < r0 + STAT_ROWS && r < M; r += 4) {
        const float v = X[r * HID + c];
        s += v; q += v * v;
    }
    sS[tid] = s; sQ[tid] = q;
    __syncthreads();
    if (tid < 64) {
        s = sS[tid] + sS[tid + 64] + sS[tid + 128] + sS[tid + 192];
        q = sQ[tid] + sQ[tid + 64] + sQ[tid + 128] + sQ[tid + 192];
        atomicAdd(&sums[tid], s);
        atomicAdd(&sumsq[tid], q);
    }
}

// Fold BN stats into per-channel scale/shift (64 threads).
__global__ void bn_prep_kernel(const float* __restrict__ sums,
                               const float* __restrict__ sumsq,
                               const float* __restrict__ g,
                               const float* __restrict__ beta,
                               float Minv, float* __restrict__ scale,
                               float* __restrict__ shift)
{
    const int c = threadIdx.x;
    if (c >= HID) return;
    const float mu  = sums[c] * Minv;
    const float var = sumsq[c] * Minv - mu * mu;
    const float sc  = g[c] * rsqrtf(var + EPS_BN);
    scale[c] = sc;
    shift[c] = beta[c] - mu * sc;
}

// base += relu(scale[c]*x + shift[c])   (float4 over [M,64] rows)
__global__ void bn_residual_kernel(float4* __restrict__ base,
                                   const float4* __restrict__ xn,
                                   const float4* __restrict__ scale,
                                   const float4* __restrict__ shift,
                                   long long n4)
{
    const long long i = (long long)blockIdx.x * blockDim.x + threadIdx.x;
    if (i >= n4) return;
    const int c4 = (int)(i & 15);
    const float4 x = xn[i], sc = scale[c4], sh = shift[c4];
    float4 b = base[i];
    b.x += fmaxf(fmaf(sc.x, x.x, sh.x), 0.f);
    b.y += fmaxf(fmaf(sc.y, x.y, sh.y), 0.f);
    b.z += fmaxf(fmaf(sc.z, x.z, sh.z), 0.f);
    b.w += fmaxf(fmaf(sc.w, x.w, sh.w), 0.f);
    base[i] = b;
}

__global__ void readout_scatter_kernel(const float* __restrict__ h,
                                       const int* __restrict__ gid,
                                       float* __restrict__ hg,
                                       float* __restrict__ counts, long long n)
{
    const long long idx = (long long)blockIdx.x * blockDim.x + threadIdx.x;
    if (idx >= n * HID) return;
    const int c       = (int)(idx & (HID - 1));
    const long long v = idx >> 6;
    const int g = gid[v];
    atomicAdd(&hg[(long long)g * HID + c], h[idx]);
    if (c == 0) atomicAdd(&counts[g], 1.0f);
}

__global__ void readout_concat_kernel(const float* __restrict__ hg,
                                      const float* __restrict__ counts,
                                      const float* __restrict__ state,
                                      const float* __restrict__ action,
                                      float* __restrict__ xcat)
{
    const int idx = blockIdx.x * blockDim.x + threadIdx.x;
    if (idx >= BG * CAT_DIM) return;
    const int b = idx / CAT_DIM, j = idx % CAT_DIM;
    float v;
    if (j < HID)                  v = hg[b * HID + j] / fmaxf(counts[b], 1.0f);
    else if (j < HID + STATE_DIM) v = state[b * STATE_DIM + (j - HID)];
    else                          v = action[b * ACT_DIM + (j - HID - STATE_DIM)];
    xcat[idx] = v;
}

__global__ __launch_bounds__(256)
void final_dot_kernel(const float* __restrict__ x, const float* __restrict__ w,
                      const float* __restrict__ b, float* __restrict__ out)
{
    __shared__ float red[256];
    const int tid = threadIdx.x;
    const int row = blockIdx.x;
    red[tid] = x[row * MLP2 + tid] * w[tid];
    __syncthreads();
    for (int s = 128; s > 0; s >>= 1) {
        if (tid < s) red[tid] += red[tid + s];
        __syncthreads();
    }
    if (tid == 0) out[row] = red[0] + b[0];
}

// ---------------------------------------------------------------------------
// Launch
// ---------------------------------------------------------------------------
static inline dim3 gemmGrid(int M, int Nout) {
    return dim3((M + TILE_M - 1) / TILE_M, Nout / TILE_N);
}
static inline unsigned tilesGrid(int M) {
    unsigned t = (unsigned)(M / 32);
    return t < 2048u ? t : 2048u;
}

extern "C" void kernel_launch(void* const* d_in, const int* in_sizes, int n_in,
                              void* d_out, int out_size, void* d_ws, size_t ws_size,
                              hipStream_t stream)
{
    const float* h_in   = (const float*)d_in[0];
    const float* e_feat = (const float*)d_in[1];
    const int*   src    = (const int*)  d_in[2];
    const int*   dst    = (const int*)  d_in[3];
    const int*   gids   = (const int*)  d_in[4];
    const float* state  = (const float*)d_in[5];
    const float* action = (const float*)d_in[6];
    const float* emb_h_W = (const float*)d_in[7];
    const float* emb_h_b = (const float*)d_in[8];
    const float* emb_e_W = (const float*)d_in[9];
    const float* emb_e_b = (const float*)d_in[10];
    const float* A_W = (const float*)d_in[11]; const float* A_b = (const float*)d_in[12];
    const float* B_W = (const float*)d_in[13]; const float* B_b = (const float*)d_in[14];
    const float* C_W = (const float*)d_in[15]; const float* C_b = (const float*)d_in[16];
    const float* D_W = (const float*)d_in[17]; const float* D_b = (const float*)d_in[18];
    const float* E_W = (const float*)d_in[19]; const float* E_b = (const float*)d_in[20];
    const float* bn_h_g = (const float*)d_in[21]; const float* bn_h_beta = (const float*)d_in[22];
    const float* bn_e_g = (const float*)d_in[23]; const float* bn_e_beta = (const float*)d_in[24];
    const float* l1_W = (const float*)d_in[25]; const float* l1_b = (const float*)d_in[26];
    const float* l2_W = (const float*)d_in[27]; const float* l2_b = (const float*)d_in[28];
    const float* l3_W = (const float*)d_in[29]; const float* l3_b = (const float*)d_in[30];
    (void)in_sizes; (void)n_in; (void)out_size; (void)ws_size;

    float* ws = (float*)d_ws;
    size_t o = 0;
    float* h    = ws + o; o += (size_t)NN * HID;
    float* e    = ws + o; o += (size_t)EE * HID;
    float* ehat = ws + o; o += (size_t)EE * HID;
    float* Ah   = ws + o; o += (size_t)NN * HID;
    float* Bh   = ws + o; o += (size_t)NN * HID;
    float* Dh   = ws + o; o += (size_t)NN * HID;
    float* Eh   = ws + o; o += (size_t)NN * HID;
    float* hnew = ws + o; o += (size_t)NN * HID;
    float* num  = ws + o; o += (size_t)NN * HID;   // num and den adjacent:
    float* den  = ws + o; o += (size_t)NN * HID;   // zeroed in one launch
    float* hg   = ws + o; o += (size_t)BG * HID;
    float* cnts = ws + o; o += (size_t)BG;
    float* xcat = ws + o; o += (size_t)BG * CAT_DIM;
    float* x1   = ws + o; o += (size_t)BG * MLP1;
    float* x2   = ws + o; o += (size_t)BG * MLP2;
    float* st   = ws + o; o += 4 * HID;   // [sum_h | sq_h | sum_e | sq_e]
    float* bnsc = ws + o; o += HID;       // BN scale
    float* bnsh = ws + o; o += HID;       // BN shift

    const long long nH = (long long)NN * HID;
    const long long nE = (long long)EE * HID;
    const dim3 blk(256);
    const dim3 grdH4((unsigned)((nH / 4 + 255) / 256));
    const dim3 grdE4((unsigned)((nE / 4 + 255) / 256));
    const dim3 grdHs((unsigned)((nH + 255) / 256));

    // ---- embeddings (generic kernel; K=6 and K=2, recip on edges) ----
    gemm_bias_wmma<<<gemmGrid(NN, HID), blk, 0, stream>>>(h_in, emb_h_W, emb_h_b, h,
                                                          NN, 6, HID, 0, 0);
    gemm_bias_wmma<<<gemmGrid(EE, HID), blk, 0, stream>>>(e_feat, emb_e_W, emb_e_b, e,
                                                          EE, 2, HID, 0, 1);

    // ---- GatedGCN layers ----
    for (int l = 0; l < NLAYER; ++l) {
        const size_t wo = (size_t)l * HID * HID;
        const size_t bo = (size_t)l * HID;

        node_gemm4_wmma<<<tilesGrid(NN), blk, 0, stream>>>(
            h, A_W + wo, A_b + bo, B_W + wo, B_b + bo,
               D_W + wo, D_b + bo, E_W + wo, E_b + bo,
            Ah, Bh, Dh, Eh, NN);

        zero4_kernel<<<256, blk, 0, stream>>>((float4*)num, 2 * nH / 4); // num+den
        zero4_kernel<<<1, blk, 0, stream>>>((float4*)st, HID);           // 4*HID floats

        edge_gemm_fused<<<tilesGrid(EE), blk, 0, stream>>>(
            e, C_W + wo, C_b + bo, src, dst, Dh, Eh, Bh,
            ehat, num, den, st + 2 * HID, st + 3 * HID, EE);

        node_update_kernel<<<grdH4, blk, 0, stream>>>(
            (const float4*)Ah, (const float4*)num, (const float4*)den,
            (float4*)hnew, nH / 4);
        stats_kernel<<<NN / STAT_ROWS, blk, 0, stream>>>(hnew, NN, st, st + HID);

        bn_prep_kernel<<<1, HID, 0, stream>>>(st, st + HID, bn_h_g + bo, bn_h_beta + bo,
                                              1.0f / NN, bnsc, bnsh);
        bn_residual_kernel<<<grdH4, blk, 0, stream>>>(
            (float4*)h, (const float4*)hnew, (const float4*)bnsc, (const float4*)bnsh,
            nH / 4);
        bn_prep_kernel<<<1, HID, 0, stream>>>(st + 2 * HID, st + 3 * HID,
                                              bn_e_g + bo, bn_e_beta + bo,
                                              1.0f / EE, bnsc, bnsh);
        bn_residual_kernel<<<grdE4, blk, 0, stream>>>(
            (float4*)e, (const float4*)ehat, (const float4*)bnsc, (const float4*)bnsh,
            nE / 4);
    }

    // ---- readout + MLP head ----
    zero4_kernel<<<64, blk, 0, stream>>>((float4*)hg, (long long)BG * HID / 4);
    zero4_kernel<<<1, blk, 0, stream>>>((float4*)cnts, BG / 4);
    readout_scatter_kernel<<<grdHs, blk, 0, stream>>>(h, gids, hg, cnts, NN);
    readout_concat_kernel<<<(BG * CAT_DIM + 255) / 256, blk, 0, stream>>>(
        hg, cnts, state, action, xcat);

    gemm_bias_wmma<<<gemmGrid(BG, MLP1), blk, 0, stream>>>(xcat, l1_W, l1_b, x1,
                                                           BG, CAT_DIM, MLP1, 1, 0);
    gemm_bias_wmma<<<gemmGrid(BG, MLP2), blk, 0, stream>>>(x1, l2_W, l2_b, x2,
                                                           BG, MLP1, MLP2, 1, 0);
    final_dot_kernel<<<BG, blk, 0, stream>>>(x2, l3_W, l3_b, (float*)d_out);
}
